// VectorQuantizer2D_57741540328090
// MI455X (gfx1250) — matrix operand order repaired and checked by reading, not compiled, not run on hardware
//
#include <hip/hip_runtime.h>
#include <hip/hip_bf16.h>

typedef __attribute__((ext_vector_type(16))) _Float16 v16h;
typedef __attribute__((ext_vector_type(8)))  _Float16 v8h;
typedef __attribute__((ext_vector_type(8)))  float    v8f;

#define KDIM      256
#define NCODES    1024
#define SD        272                    // padded LDS row stride (halfs): 544B, 16B-aligned
#define MTILE     128                    // rows per workgroup
#define NCHUNK    128                    // codebook entries per LDS chunk
#define BUF_HALFS (MTILE * SD)           // 34816 halfs = 69632 B per buffer
#define NROWS_TOT 65536                  // 64 * 32 * 32
#define ZQ_ELEMS  16777216               // 64 * 256 * 32 * 32

// ---- CDNA5 async global->LDS path (ASYNCcnt), with safe fallback ----------
#if defined(__HIP_DEVICE_COMPILE__) && defined(__AMDGCN__) && \
    __has_builtin(__builtin_amdgcn_global_load_async_to_lds_b128)
#define HAVE_ASYNC_COPY 1
#else
#define HAVE_ASYNC_COPY 0
#endif

#if defined(__HIP_DEVICE_COMPILE__) && defined(__AMDGCN__) && \
    __has_builtin(__builtin_amdgcn_s_wait_asynccnt)
#define WAIT_ASYNC0() __builtin_amdgcn_s_wait_asynccnt(0)
#elif HAVE_ASYNC_COPY
#define WAIT_ASYNC0() asm volatile("s_wait_asynccnt 0x0" ::: "memory")
#else
#define WAIT_ASYNC0() ((void)0)
#endif

extern __shared__ char smem_raw[];

// ---------------------------------------------------------------------------
// K0: codebook fp32 -> f16, ||e||^2 per code, zero the loss accumulator
// ---------------------------------------------------------------------------
__global__ __launch_bounds__(256) void vq_prep(const float* __restrict__ emb,
                                               _Float16* __restrict__ embH,
                                               float* __restrict__ normE,
                                               float* __restrict__ lossAcc) {
  __shared__ float red[256];
  const int c = blockIdx.x;
  const int t = threadIdx.x;
  float v = emb[c * KDIM + t];
  embH[c * KDIM + t] = (_Float16)v;
  red[t] = v * v;
  __syncthreads();
  for (int s = 128; s > 0; s >>= 1) {
    if (t < s) red[t] += red[t + s];
    __syncthreads();
  }
  if (t == 0) {
    normE[c] = red[0];
    if (c == 0) lossAcc[0] = 0.0f;
  }
}

// Issue one 128-code chunk copy (64 KB) embH -> LDS buffer.
// Async path: no VGPR staging; tracked by ASYNCcnt, drained by WAIT_ASYNC0().
// Fallback: plain load/store (completes by the ds stores themselves).
__device__ __forceinline__ void chunk_copy_issue(const _Float16* __restrict__ src,
                                                 _Float16* dst, int tid) {
#if HAVE_ASYNC_COPY
  typedef int v4i_ __attribute__((vector_size(16)));
  typedef __attribute__((address_space(1))) v4i_ gv4i;   // prints as "int4 __device__*"
  typedef __attribute__((address_space(3))) v4i_ lv4i;   // prints as "int4 __shared__*"
  _Float16* snc = const_cast<_Float16*>(src);
  #pragma unroll 4
  for (int it = 0; it < 32; ++it) {
    int L = tid + 128 * it;              // 0..4095 16-byte units
    int code = L >> 5, seg = L & 31;
    __builtin_amdgcn_global_load_async_to_lds_b128(
        (gv4i*)(snc + code * KDIM + seg * 8),
        (lv4i*)(dst + code * SD + seg * 8), 0, 0);
  }
#else
  const v8h* s = (const v8h*)src;
  #pragma unroll 4
  for (int it = 0; it < 32; ++it) {
    int L = tid + 128 * it;
    int code = L >> 5, seg = L & 31;
    *(v8h*)(dst + code * SD + seg * 8) = s[code * 32 + seg];
  }
#endif
}

// ---------------------------------------------------------------------------
// K1: fused GEMM (f16 WMMA, f32 accum) + argmin + row loss
//   grid = 512 blocks of 128 threads (4 wave32). Wave w owns 32 rows; its A
//   fragments (32 rows x 256 K, f16) live entirely in registers (128 VGPRs)
//   so each B fragment read from LDS feeds two v_wmma_f32_16x16x32_f16.
// ---------------------------------------------------------------------------
__global__ __launch_bounds__(128, 1) void vq_main(const float* __restrict__ z,
                                                  const _Float16* __restrict__ embH,
                                                  const float* __restrict__ normE,
                                                  int* __restrict__ idxOut,
                                                  float* __restrict__ lossAcc) {
  _Float16* buf0 = (_Float16*)smem_raw;
  _Float16* buf1 = buf0 + BUF_HALFS;
  float* znormLds = (float*)(smem_raw + (size_t)2 * BUF_HALFS * sizeof(_Float16));

  const int tid  = threadIdx.x;
  const int lane = tid & 31;
  const int wv   = tid >> 5;      // wave 0..3
  const int l15  = lane & 15;
  const int hiH  = lane >> 4;     // 0 for lanes 0-15, 1 for lanes 16-31

  const int wgRow0 = blockIdx.x * MTILE;      // flat row = n*1024 + hw
  const int n      = wgRow0 >> 10;
  const int hwBase = wgRow0 & 1023;
  const float* zb  = z + (size_t)n * (KDIM * 1024) + hwBase;

  // ---- Phase 1b first: E chunk 0 -> buf1 (async copy overlaps z staging) ----
  chunk_copy_issue(embH, buf1, tid);

  // ---- Phase 1a: stage Z tile into buf0 transposed -> zt[hw][d] as f16 ----
  // z is (N, D, HW): for each d, 128 contiguous floats -> coalesced.
  #pragma unroll 4
  for (int it = 0; it < 64; ++it) {
    int L4  = tid + 128 * it;          // 0..8191 float4s
    int d   = L4 >> 5;                 // 32 float4 per d-row of 128 floats
    int hw4 = (L4 & 31) << 2;
    const float4 v = *(const float4*)(zb + (size_t)d * 1024 + hw4);
    _Float16* p = buf0 + d;
    p[(hw4 + 0) * SD] = (_Float16)v.x;
    p[(hw4 + 1) * SD] = (_Float16)v.y;
    p[(hw4 + 2) * SD] = (_Float16)v.z;
    p[(hw4 + 3) * SD] = (_Float16)v.w;
  }
  WAIT_ASYNC0();
  __syncthreads();

  // ---- Phase 2a: ||z_row||^2 from the staged tile (one row per thread) ----
  {
    float s = 0.0f;
    const v8h* r = (const v8h*)(buf0 + tid * SD);
    #pragma unroll 4
    for (int seg = 0; seg < 32; ++seg) {
      v8h h = r[seg];
      #pragma unroll
      for (int j = 0; j < 8; ++j) { float x = (float)h[j]; s += x * x; }
    }
    znormLds[tid] = s;
  }

  // ---- Phase 2b: A fragments (two 16-row sets) -> registers ----
  // 16-bit A 16x32 layout: lanes 0-15 hold K {0-7,16-23}, lanes 16-31 K {8-15,24-31}
  v16h a0[8], a1[8];
  {
    const _Float16* r0 = buf0 + (wv * 32 + l15) * SD;
    const _Float16* r1 = r0 + 16 * SD;
    #pragma unroll
    for (int kk = 0; kk < 8; ++kk) {
      int kb = kk * 32 + hiH * 8;
      v8h lo0 = *(const v8h*)(r0 + kb);
      v8h hi0 = *(const v8h*)(r0 + kb + 16);
      v8h lo1 = *(const v8h*)(r1 + kb);
      v8h hi1 = *(const v8h*)(r1 + kb + 16);
      #pragma unroll
      for (int j = 0; j < 8; ++j) {
        a0[kk][j] = lo0[j]; a0[kk][8 + j] = hi0[j];
        a1[kk][j] = lo1[j]; a1[kk][8 + j] = hi1[j];
      }
    }
  }
  __syncthreads();                     // buf0 free for reuse as E buffer

  float minv0[8], minv1[8];
  int   mini0[8], mini1[8];
  #pragma unroll
  for (int v = 0; v < 8; ++v) {
    minv0[v] = 3.0e38f; minv1[v] = 3.0e38f; mini0[v] = 0; mini1[v] = 0;
  }

  // ---- Main loop: 8 chunks of 128 codes, double-buffered through LDS ----
  _Float16* cur = buf1;
  _Float16* nxt = buf0;
  for (int ch = 0; ch < 8; ++ch) {
    if (ch < 7)                        // async prefetch next chunk into other buffer
      chunk_copy_issue(embH + (size_t)(ch + 1) * (NCHUNK * KDIM), nxt, tid);

    // 8 column tiles of 16 codes; per tile: 8 B-fragments feed 2 WMMAs each
    for (int ct = 0; ct < 8; ++ct) {
      // 16-bit B 32x16 layout: lane=col, lanes 0-15 K {0-7,16-23}, lanes 16-31 K {8-15,24-31}
      const _Float16* bcol = cur + (ct * 16 + l15) * SD + hiH * 8;
      v8f c0, c1;
      #pragma unroll
      for (int v = 0; v < 8; ++v) { c0[v] = 0.0f; c1[v] = 0.0f; }
      #pragma unroll
      for (int kk = 0; kk < 8; ++kk) {
        v8h lo = *(const v8h*)(bcol + kk * 32);
        v8h hi = *(const v8h*)(bcol + kk * 32 + 16);
        v16h b;
        #pragma unroll
        for (int j = 0; j < 8; ++j) { b[j] = lo[j]; b[8 + j] = hi[j]; }
        c0 = __builtin_amdgcn_wmma_f32_16x16x32_f16(false, a0[kk], false, b,
                                                    (short)0, c0, false, false);
        c1 = __builtin_amdgcn_wmma_f32_16x16x32_f16(false, a1[kk], false, b,
                                                    (short)0, c1, false, false);
      }
      int colG = ch * 128 + ct * 16 + l15;
      float ne = normE[colG];
      #pragma unroll
      for (int v = 0; v < 8; ++v) {
        float s0 = fmaf(-2.0f, c0[v], ne);   // ||e||^2 - 2 z.e  (row-const ||z||^2 dropped)
        if (s0 < minv0[v]) { minv0[v] = s0; mini0[v] = colG; }
        float s1 = fmaf(-2.0f, c1[v], ne);
        if (s1 < minv1[v]) { minv1[v] = s1; mini1[v] = colG; }
      }
    }
    WAIT_ASYNC0();                     // drain this wave's async copies into nxt
    __syncthreads();                   // all waves done reading cur / writing nxt
    _Float16* t = cur; cur = nxt; nxt = t;
  }

  // ---- Argmin reduction across the 16 lanes sharing each row ----
  float lossPart = 0.0f;
  #pragma unroll
  for (int v = 0; v < 8; ++v) {
    float m0 = minv0[v]; int i0 = mini0[v];
    float m1 = minv1[v]; int i1 = mini1[v];
    #pragma unroll
    for (int off = 8; off > 0; off >>= 1) {   // xor masks < 16: stays in 16-lane halves
      float om0 = __shfl_xor(m0, off, 32); int oi0 = __shfl_xor(i0, off, 32);
      if (om0 < m0 || (om0 == m0 && oi0 < i0)) { m0 = om0; i0 = oi0; }
      float om1 = __shfl_xor(m1, off, 32); int oi1 = __shfl_xor(i1, off, 32);
      if (om1 < m1 || (om1 == m1 && oi1 < i1)) { m1 = om1; i1 = oi1; }
    }
    if (l15 == 0) {
      int rl0 = wv * 32 + hiH * 8 + v;        // C-layout: row = vgpr + 8*(lane>=16)
      int rl1 = rl0 + 16;
      idxOut[wgRow0 + rl0] = i0;
      idxOut[wgRow0 + rl1] = i1;
      lossPart += (znormLds[rl0] + m0) + (znormLds[rl1] + m1);
    }
  }
  if (l15 == 0) atomicAdd(lossAcc, lossPart);
}

// ---------------------------------------------------------------------------
// K2: z_q gather -> (N, D, H, W), float4 stores; codebook gathers hit L2
// ---------------------------------------------------------------------------
__global__ __launch_bounds__(256) void vq_gather(const float* __restrict__ emb,
                                                 const int* __restrict__ idx,
                                                 float* __restrict__ out) {
  int f4  = blockIdx.x * 256 + threadIdx.x;   // 0 .. 4194303
  int nd  = f4 >> 8;
  int hw4 = (f4 & 255) << 2;
  int n   = nd >> 8;
  int d   = nd & 255;
  const int4 iv = *(const int4*)(idx + n * 1024 + hw4);
  float4 o;
  o.x = emb[(size_t)iv.x * KDIM + d];
  o.y = emb[(size_t)iv.y * KDIM + d];
  o.z = emb[(size_t)iv.z * KDIM + d];
  o.w = emb[(size_t)iv.w * KDIM + d];
  *(float4*)(out + (size_t)f4 * 4) = o;
}

// ---------------------------------------------------------------------------
// K3: indices (as float) + final loss scalar
// ---------------------------------------------------------------------------
__global__ __launch_bounds__(256) void vq_finish(const int* __restrict__ idx,
                                                 const float* __restrict__ lossAcc,
                                                 float* __restrict__ out) {
  int i = blockIdx.x * 256 + threadIdx.x;
  out[ZQ_ELEMS + i] = (float)idx[i];
  if (i == 0)
    out[ZQ_ELEMS + NROWS_TOT] = 1.02f * lossAcc[0] * (1.0f / 16777216.0f);
}

// ---------------------------------------------------------------------------
extern "C" void kernel_launch(void* const* d_in, const int* in_sizes, int n_in,
                              void* d_out, int out_size, void* d_ws, size_t ws_size,
                              hipStream_t stream) {
  (void)in_sizes; (void)n_in; (void)out_size; (void)ws_size;
  const float* z   = (const float*)d_in[0];
  const float* emb = (const float*)d_in[1];
  float* out = (float*)d_out;
  char* ws = (char*)d_ws;

  _Float16* embH  = (_Float16*)ws;                 // 524288 B
  float*    normE = (float*)(ws + 524288);         //   4096 B
  int*      idxOut= (int*)(ws + 528384);           // 262144 B
  float*    lossAcc = (float*)(ws + 790528);       //      4 B

  vq_prep<<<NCODES, 256, 0, stream>>>(emb, embH, normE, lossAcc);

  size_t smem = (size_t)2 * BUF_HALFS * sizeof(_Float16) + MTILE * sizeof(float);
  vq_main<<<NROWS_TOT / MTILE, 128, smem, stream>>>(z, embH, normE, idxOut, lossAcc);

  vq_gather<<<ZQ_ELEMS / 1024, 256, 0, stream>>>(emb, idxOut, out);
  vq_finish<<<NROWS_TOT / 256, 256, 0, stream>>>(idxOut, lossAcc, out);
}
